// WindowAttention_29575144801074
// MI455X (gfx1250) — compile-verified, matrix operand order checked by reading, not run on hardware
//
#include <hip/hip_runtime.h>
#include <hip/hip_bf16.h>

// ---------------- CDNA5 WMMA types ----------------
typedef __attribute__((ext_vector_type(16))) _Float16 v16h;
typedef __attribute__((ext_vector_type(8)))  _Float16 v8h;
typedef __attribute__((ext_vector_type(4)))  _Float16 v4h;
typedef __attribute__((ext_vector_type(8)))  float    v8f;

#define NTOK   49      // tokens per window
#define CDIM   256     // channels
#define NHEAD  8
#define HDIM   32      // head dim
#define NWIN   64      // windows per image (mask count)
#define NBATCH 4096
#define QSCALE 0.17677669529663687f  // 32^-0.5

// Per-head LDS slice (halves):
//   qh  : 49x32  = 1568
//   kh  : 49x32  = 1568
//   vth : 32x64  = 2048   (v transposed: [d][key], keys 49..63 zeroed)
//   ph  : 49x64  = 3136   (aliases qh+kh exactly)
#define HEAD_LDS 5184
#define XS_HALVES (NTOK * CDIM)                       // 12544 (x tile; reused as proj input)
#define SMEM_HALVES (NHEAD * HEAD_LDS + XS_HALVES)    // 54016 halves = 108032 bytes

#define QKV_W_ELEMS (3 * CDIM * CDIM)                 // 196608
#define PROJ_W_ELEMS (CDIM * CDIM)                    // 65536
#define W16_ELEMS (QKV_W_ELEMS + PROJ_W_ELEMS)        // 262144

__device__ __forceinline__ v8f vzero8() {
  v8f z;
#pragma unroll
  for (int i = 0; i < 8; ++i) z[i] = 0.0f;
  return z;
}

// 16x32 f16 fragment (A layout; same pattern serves B^T rows) from f32 memory.
__device__ __forceinline__ v16h frag_from_f32(const float* __restrict__ p) {
  const float4* q = (const float4*)p;
  float4 a0 = q[0], a1 = q[1];   // K kb..kb+7
  float4 b0 = q[4], b1 = q[5];   // K kb+16..kb+23
  v16h a;
  a[0] = (_Float16)a0.x; a[1] = (_Float16)a0.y; a[2] = (_Float16)a0.z; a[3] = (_Float16)a0.w;
  a[4] = (_Float16)a1.x; a[5] = (_Float16)a1.y; a[6] = (_Float16)a1.z; a[7] = (_Float16)a1.w;
  a[8] = (_Float16)b0.x; a[9] = (_Float16)b0.y; a[10] = (_Float16)b0.z; a[11] = (_Float16)b0.w;
  a[12] = (_Float16)b1.x; a[13] = (_Float16)b1.y; a[14] = (_Float16)b1.z; a[15] = (_Float16)b1.w;
  return a;
}

// 16x32 f16 fragment from f16 memory (global or LDS): two 16B chunks.
__device__ __forceinline__ v16h frag_h16(const _Float16* p) {
  v8h lo = *(const v8h*)p;
  v8h hi = *(const v8h*)(p + 16);
  v16h a;
#pragma unroll
  for (int i = 0; i < 8; ++i) { a[i] = lo[i]; a[i + 8] = hi[i]; }
  return a;
}

__device__ __forceinline__ v16h frag_lds(const _Float16* base, int ld,
                                         int rt, int kt, int lane, int rclamp) {
  int m = rt * 16 + (lane & 15);
  m = (m > rclamp) ? rclamp : m;          // clamp keeps reads in-bounds; outputs ignored
  int kb = kt * 32 + ((lane >> 4) << 3);
  return frag_h16(base + m * ld + kb);
}

__device__ __forceinline__ v8f wmma16(v16h a, v16h b, v8f c) {
  return __builtin_amdgcn_wmma_f32_16x16x32_f16(false, a, false, b, (short)0, c,
                                                false, false);
}

// ---------------- pre-pass: weights f32 -> f16 in workspace ----------------
__global__ __launch_bounds__(256)
void convert_weights_kernel(const float* __restrict__ qkv_w,
                            const float* __restrict__ proj_w,
                            _Float16* __restrict__ wbuf) {
  int i = blockIdx.x * 256 + threadIdx.x;            // grid covers W16_ELEMS
  float v = (i < QKV_W_ELEMS) ? qkv_w[i] : proj_w[i - QKV_W_ELEMS];
  wbuf[i] = (_Float16)v;
}

// ---------------- fused window attention ----------------
template <bool WF16>
__global__ __launch_bounds__(256, 2)
void swin_window_attn_kernel(const float* __restrict__ x,          // (4096,49,256)
                             const float* __restrict__ mask,       // (64,49,49)
                             const float* __restrict__ qkv_w,      // (768,256)
                             const float* __restrict__ qkv_b,      // (768)
                             const float* __restrict__ proj_w,     // (256,256)
                             const float* __restrict__ proj_b,     // (256)
                             const float* __restrict__ bias_table, // (169,8)
                             const int*   __restrict__ rel_index,  // (49,49)
                             const _Float16* __restrict__ w16,     // f16 weights or null
                             float* __restrict__ out)              // (4096,49,256)
{
  extern __shared__ _Float16 smem[];

  const int tid  = threadIdx.x;
  const int lane = tid & 31;
  const int h    = tid >> 5;              // wave id == head id (8 waves)
  const int b    = blockIdx.x;            // window index
  const int win  = b & (NWIN - 1);        // mask slot (b % 64)
  const int nlo  = lane & 15;             // N coordinate inside a tile
  const int rhi  = (lane >> 4) << 3;      // +8 row offset for upper half-wave

  _Float16* qh  = smem + h * HEAD_LDS;                 // 49x32
  _Float16* kh  = qh + NTOK * HDIM;                    // 49x32
  _Float16* vth = kh + NTOK * HDIM;                    // 32x64 (v^T)
  _Float16* ph  = qh;                                  // 49x64, aliases qh+kh
  _Float16* xs  = smem + NHEAD * HEAD_LDS;             // 49x256 f16 x tile
  _Float16* outs = xs;                                 // proj input reuses xs region

  const float* xw    = x + (size_t)b * NTOK * CDIM;
  const float* maskw = mask + (size_t)win * NTOK * NTOK;
  const _Float16* wq16 = w16;                          // (768,256) f16
  const _Float16* wp16 = w16 + QKV_W_ELEMS;            // (256,256) f16

  // Warm L2 for the weights (global_prefetch_b8 path).
  if (WF16) {
    __builtin_prefetch(w16 + tid * (W16_ELEMS / 256), 0, 1);
  } else {
    __builtin_prefetch(qkv_w + tid * (QKV_W_ELEMS / 256), 0, 1);
    __builtin_prefetch(proj_w + tid * (PROJ_W_ELEMS / 256), 0, 1);
  }

  // Zero my head's v^T slice so padded key columns 49..63 contribute 0.
  for (int i = lane; i < HDIM * 64; i += 32) vth[i] = (_Float16)0.0f;

  // ---------------- Phase 0: x tile f32 -> f16 into LDS (once) ----------------
  {
    const float4* xv = (const float4*)xw;             // 3136 float4s
    for (int i = tid; i < (NTOK * CDIM) / 4; i += 256) {
      float4 t = xv[i];
      v4h hv;
      hv[0] = (_Float16)t.x; hv[1] = (_Float16)t.y;
      hv[2] = (_Float16)t.z; hv[3] = (_Float16)t.w;
      *(v4h*)(xs + i * 4) = hv;                       // ds_store_b64
    }
  }
  __syncthreads();

  // ---------------- Phase 1: QKV projection for this head ----------------
#pragma unroll
  for (int which = 0; which < 3; ++which) {           // 0=q 1=k 2=v
#pragma unroll
    for (int nt = 0; nt < 2; ++nt) {
      const int ncb = which * 256 + h * HDIM + nt * 16;
      v8f acc[4];
#pragma unroll
      for (int mt = 0; mt < 4; ++mt) acc[mt] = vzero8();

      for (int kt = 0; kt < 8; ++kt) {                // K = 256
        const int woff = (ncb + nlo) * CDIM + kt * 32 + rhi;
        v16h bw = WF16 ? frag_h16(wq16 + woff) : frag_from_f32(qkv_w + woff);
#pragma unroll
        for (int mt = 0; mt < 4; ++mt) {
          v16h a = frag_lds(xs, CDIM, mt, kt, lane, NTOK - 1);
          acc[mt] = wmma16(a, bw, acc[mt]);
        }
      }

      const float bias = qkv_b[ncb + nlo];
      const int d = nt * 16 + nlo;                    // head-local dim
#pragma unroll
      for (int mt = 0; mt < 3; ++mt) {                // rows 0..47: always valid
#pragma unroll
        for (int r = 0; r < 8; ++r) {
          const int row = mt * 16 + r + rhi;
          float v = acc[mt][r] + bias;
          if (which == 0)      qh[row * HDIM + d] = (_Float16)(v * QSCALE);
          else if (which == 1) kh[row * HDIM + d] = (_Float16)v;
          else                 vth[d * 64 + row]  = (_Float16)v;
        }
      }
      if (lane < 16) {                                // mt=3, r=0 -> row 48
        float v = acc[3][0] + bias;
        if (which == 0)      qh[48 * HDIM + d] = (_Float16)(v * QSCALE);
        else if (which == 1) kh[48 * HDIM + d] = (_Float16)v;
        else                 vth[d * 64 + 48]  = (_Float16)v;
      }
    }
  }

  // ---------------- Phase 2: S = q @ k^T  (49x64, keys padded) ----------------
  v8f sacc[4][4];
#pragma unroll
  for (int mt = 0; mt < 4; ++mt)
#pragma unroll
    for (int nt = 0; nt < 4; ++nt) sacc[mt][nt] = vzero8();

  {
    v16h aq[4];
#pragma unroll
    for (int mt = 0; mt < 4; ++mt) aq[mt] = frag_lds(qh, HDIM, mt, 0, lane, NTOK - 1);
#pragma unroll
    for (int nt = 0; nt < 4; ++nt) {
      v16h bk = frag_lds(kh, HDIM, nt, 0, lane, NTOK - 1);
#pragma unroll
      for (int mt = 0; mt < 4; ++mt) sacc[mt][nt] = wmma16(aq[mt], bk, sacc[mt][nt]);
    }
  }

  // Bias + mask + softmax (row lives in one 16-lane half per C element).
  // Branch-free gathers: clamped indices + selects; stores specialized so only
  // the single real row-48 element needs a lane predicate.
  auto process_row = [&](int mt, int r, bool tail) {
    const int row  = mt * 16 + r + rhi;
    const int rowc = (row > NTOK - 1) ? (NTOK - 1) : row;   // only tail can exceed
    const int gbase = rowc * NTOK;
#pragma unroll
    for (int nt = 0; nt < 4; ++nt) {
      const int key  = nt * 16 + nlo;
      const int keyc = (nt == 3) ? ((nlo == 0) ? 48 : 48 - 1 + (nlo == 0)) : key;
      // simpler clamp: nt<3 -> key valid; nt==3 -> clamp to 48
      const int kidx = (nt == 3) ? 48 : key;
      int   ridx = rel_index[gbase + kidx];
      float addv = bias_table[ridx * NHEAD + h] + maskw[gbase + kidx];
      float s = sacc[mt][nt][r] + addv;
      if (nt == 3) s = (nlo == 0) ? s : -1e30f;             // padded keys 49..63
      sacc[mt][nt][r] = s;
      (void)keyc;
    }
    float mx = fmaxf(fmaxf(sacc[mt][0][r], sacc[mt][1][r]),
                     fmaxf(sacc[mt][2][r], sacc[mt][3][r]));
    mx = fmaxf(mx, __shfl_xor(mx, 1));
    mx = fmaxf(mx, __shfl_xor(mx, 2));
    mx = fmaxf(mx, __shfl_xor(mx, 4));
    mx = fmaxf(mx, __shfl_xor(mx, 8));
    float sum = 0.0f;
#pragma unroll
    for (int nt = 0; nt < 4; ++nt) {
      float e = __expf(sacc[mt][nt][r] - mx);
      sacc[mt][nt][r] = e;
      sum += e;
    }
    sum += __shfl_xor(sum, 1);
    sum += __shfl_xor(sum, 2);
    sum += __shfl_xor(sum, 4);
    sum += __shfl_xor(sum, 8);
    const float inv = 1.0f / sum;
    if (!tail) {
#pragma unroll
      for (int nt = 0; nt < 4; ++nt)
        ph[row * 64 + nt * 16 + nlo] = (_Float16)(sacc[mt][nt][r] * inv);
    } else if (lane < 16) {                                 // row 48 only
#pragma unroll
      for (int nt = 0; nt < 4; ++nt)
        ph[48 * 64 + nt * 16 + nlo] = (_Float16)(sacc[mt][nt][r] * inv);
    }
  };

#pragma unroll
  for (int mt = 0; mt < 3; ++mt)
#pragma unroll
    for (int r = 0; r < 8; ++r) process_row(mt, r, false);
  process_row(3, 0, true);                                  // row 48 (+ garbage half)

  // ---------------- Phase 3: O_h = P @ V  (49x32) ----------------
  v8f oacc[4][2];
#pragma unroll
  for (int mt = 0; mt < 4; ++mt)
#pragma unroll
    for (int nt = 0; nt < 2; ++nt) oacc[mt][nt] = vzero8();

  for (int kt = 0; kt < 2; ++kt) {                          // K = 64 padded keys
    v16h ap[4];
#pragma unroll
    for (int mt = 0; mt < 4; ++mt) ap[mt] = frag_lds(ph, 64, mt, kt, lane, NTOK - 1);
#pragma unroll
    for (int nt = 0; nt < 2; ++nt) {
      v16h bv = frag_lds(vth, 64, nt, kt, lane, HDIM - 1);
#pragma unroll
      for (int mt = 0; mt < 4; ++mt) oacc[mt][nt] = wmma16(ap[mt], bv, oacc[mt][nt]);
    }
  }

  __syncthreads();   // all waves done with per-head LDS + xs before overlaying outs

#pragma unroll
  for (int nt = 0; nt < 2; ++nt) {
    const int col = h * HDIM + nt * 16 + nlo;               // channel = head*32 + d
#pragma unroll
    for (int mt = 0; mt < 3; ++mt)
#pragma unroll
      for (int r = 0; r < 8; ++r)
        outs[(mt * 16 + r + rhi) * CDIM + col] = (_Float16)oacc[mt][nt][r];
    if (lane < 16) outs[48 * CDIM + col] = (_Float16)oacc[3][nt][0];
  }

  __syncthreads();   // outs (49x256) fully assembled

  // ---------------- Phase 4: out = O @ proj_w^T + proj_b ----------------
  v8f pacc[4][2];
#pragma unroll
  for (int mt = 0; mt < 4; ++mt)
#pragma unroll
    for (int nt = 0; nt < 2; ++nt) pacc[mt][nt] = vzero8();

  for (int kt = 0; kt < 8; ++kt) {                          // K = 256
    v16h ao[4];
#pragma unroll
    for (int mt = 0; mt < 4; ++mt) ao[mt] = frag_lds(outs, CDIM, mt, kt, lane, NTOK - 1);
#pragma unroll
    for (int nt = 0; nt < 2; ++nt) {
      const int woff = (h * HDIM + nt * 16 + nlo) * CDIM + kt * 32 + rhi;
      v16h bw = WF16 ? frag_h16(wp16 + woff) : frag_from_f32(proj_w + woff);
#pragma unroll
      for (int mt = 0; mt < 4; ++mt) pacc[mt][nt] = wmma16(ao[mt], bw, pacc[mt][nt]);
    }
  }

  float* outg = out + (size_t)b * NTOK * CDIM;
#pragma unroll
  for (int nt = 0; nt < 2; ++nt) {
    const int col = h * HDIM + nt * 16 + nlo;
    const float pb = proj_b[col];
#pragma unroll
    for (int mt = 0; mt < 3; ++mt)
#pragma unroll
      for (int r = 0; r < 8; ++r)
        outg[(mt * 16 + r + rhi) * CDIM + col] = pacc[mt][nt][r] + pb;
    if (lane < 16) outg[48 * CDIM + col] = pacc[3][nt][0] + pb;
  }
}

extern "C" void kernel_launch(void* const* d_in, const int* in_sizes, int n_in,
                              void* d_out, int out_size, void* d_ws, size_t ws_size,
                              hipStream_t stream) {
  const float* x          = (const float*)d_in[0];
  const float* mask       = (const float*)d_in[1];
  const float* qkv_w      = (const float*)d_in[2];
  const float* qkv_b      = (const float*)d_in[3];
  const float* proj_w     = (const float*)d_in[4];
  const float* proj_b     = (const float*)d_in[5];
  const float* bias_table = (const float*)d_in[6];
  const int*   rel_index  = (const int*)d_in[7];
  float* out = (float*)d_out;

  dim3 grid(NBATCH);
  dim3 block(256);
  size_t smem_bytes = (size_t)SMEM_HALVES * 2;              // 108,032 B dynamic LDS

  const bool wf16 = ws_size >= (size_t)W16_ELEMS * 2;       // fixed per-run -> deterministic
  if (wf16) {
    _Float16* wbuf = (_Float16*)d_ws;
    convert_weights_kernel<<<dim3(W16_ELEMS / 256), block, 0, stream>>>(qkv_w, proj_w, wbuf);
    swin_window_attn_kernel<true><<<grid, block, smem_bytes, stream>>>(
        x, mask, qkv_w, qkv_b, proj_w, proj_b, bias_table, rel_index, wbuf, out);
  } else {
    swin_window_attn_kernel<false><<<grid, block, smem_bytes, stream>>>(
        x, mask, qkv_w, qkv_b, proj_w, proj_b, bias_table, rel_index, nullptr, out);
  }
}
